// MHA_35227321761933
// MI455X (gfx1250) — compile-verified, matrix operand order checked by reading, not run on hardware
//
#include <hip/hip_runtime.h>
#include <hip/hip_bf16.h>
#include <math.h>

typedef __attribute__((ext_vector_type(8)))  _Float16 v8h;
typedef __attribute__((ext_vector_type(16))) _Float16 v16h;
typedef __attribute__((ext_vector_type(8)))  float    v8f;

#define S_LEN 2048
#define DIM_  4096
#define NH_   32
#define NKV_  8
#define HD_   128
#define KVD_  1024   // NKV_*HD_

union Frag16 { v16h v; v8h h[2]; };

// Load a 16x32 f16 WMMA fragment (A layout; B uses the same per-lane pattern on
// the row-major [N][K] weight matrix, i.e. column n of B == row n of W).
// Per ISA 7.12.2: lane L holds row (L&15); chunks K = 8*(L>>4)+[0..7] in v0-3
// and K = 16+8*(L>>4)+[0..7] in v4-7 -> two 16-byte loads per lane.
__device__ __forceinline__ v16h load_frag(const _Float16* base, int row, int ldk,
                                          int k0, int lane) {
  const int r    = lane & 15;
  const int half = lane >> 4;
  const _Float16* p = base + (size_t)(row + r) * (size_t)ldk + (k0 + half * 8);
  Frag16 f;
  f.h[0] = *(const v8h*)(p);
  f.h[1] = *(const v8h*)(p + 16);
  return f.v;
}

__device__ __forceinline__ v8f wmma16(v16h a, v16h b, v8f c) {
  return __builtin_amdgcn_wmma_f32_16x16x32_f16(false, a, false, b, (short)0, c,
                                                false, false);
}

__global__ void __launch_bounds__(256) cvt_f32_f16(const float* __restrict__ in,
                                                   _Float16* __restrict__ out,
                                                   size_t n) {
  for (size_t i = (size_t)blockIdx.x * blockDim.x + threadIdx.x; i < n;
       i += (size_t)gridDim.x * blockDim.x)
    out[i] = (_Float16)in[i];
}

// C[m][n] = sum_k A[m][k] * B[n][k]   (A: MxK f16 row-major, B: NxK f16 row-major)
// Block = 8 waves (2x4) computing a 128x256 tile; wave tile = 64x64 (4x4 WMMA
// tiles): 16 b128 loads feed 16 WMMAs per 32-deep K step (1.0 loads/WMMA).
// N,K are compile-time so all fragment addresses fold into one base pointer
// plus 24-bit immediate offsets (row stride K*2 = 8 KB << +-8 MB range).
template <int N, int K>
__global__ void __launch_bounds__(256) gemm_wmma_f16(const _Float16* __restrict__ A,
                                                     const _Float16* __restrict__ B,
                                                     float* __restrict__ C) {
  const int lane = threadIdx.x & 31;
  const int wave = threadIdx.x >> 5;
  const int m0 = blockIdx.y * 128 + (wave >> 2) * 64;   // 2 waves down M
  const int n0 = blockIdx.x * 256 + (wave & 3) * 64;    // 4 waves across N

  const v8f z = {0.f, 0.f, 0.f, 0.f, 0.f, 0.f, 0.f, 0.f};
  v8f acc[4][4];
#pragma unroll
  for (int i = 0; i < 4; ++i)
#pragma unroll
    for (int j = 0; j < 4; ++j) acc[i][j] = z;

  const int r    = lane & 15;
  const int half = lane >> 4;
  const _Float16* pa = A + (size_t)(m0 + r) * K + half * 8;
  const _Float16* pb = B + (size_t)(n0 + r) * K + half * 8;

  for (int k0 = 0; k0 < K; k0 += 32) {
    v16h a[4];
#pragma unroll
    for (int i = 0; i < 4; ++i) {
      Frag16 f;
      f.h[0] = *(const v8h*)(pa + (size_t)i * 16 * K + k0);
      f.h[1] = *(const v8h*)(pa + (size_t)i * 16 * K + k0 + 16);
      a[i] = f.v;
    }
#pragma unroll
    for (int j = 0; j < 4; ++j) {
      Frag16 f;
      f.h[0] = *(const v8h*)(pb + (size_t)j * 16 * K + k0);
      f.h[1] = *(const v8h*)(pb + (size_t)j * 16 * K + k0 + 16);
#pragma unroll
      for (int i = 0; i < 4; ++i)
        acc[i][j] = wmma16(a[i], f.v, acc[i][j]);
    }
  }

  const int nlo = lane & 15;
#pragma unroll
  for (int i = 0; i < 4; ++i)
#pragma unroll
    for (int j = 0; j < 4; ++j)
#pragma unroll
      for (int rr = 0; rr < 8; ++rr) {
        int m = m0 + i * 16 + rr + 8 * half;  // C-frag: VGPR rr -> row rr+8*half
        int n = n0 + j * 16 + nlo;            // lane -> column
        C[(size_t)m * N + n] = acc[i][j][rr];
      }
}

// Interleaved RoPE (pairs at 2i,2i+1) + fp32 -> f16.
__global__ void __launch_bounds__(256) rope_f16(const float* __restrict__ X,
                                                _Float16* __restrict__ Y,
                                                int nheads) {
  size_t gid = (size_t)blockIdx.x * 256 + threadIdx.x;  // over S*nheads*64
  int    i   = (int)(gid & 63);
  size_t tmp = gid >> 6;
  int    h   = (int)(tmp % nheads);
  int    t   = (int)(tmp / nheads);
  // inv_freq = theta^(-2i/128), ln(500000) = 13.122363...
  float inv_freq = __expf(-(float)i * (2.0f / 128.0f) * 13.122363377404328f);
  float ang = (float)t * inv_freq;
  float c = __cosf(ang), s = __sinf(ang);
  size_t base = ((size_t)t * nheads + h) * HD_ + 2 * i;
  float xe = X[base], xo = X[base + 1];
  Y[base]     = (_Float16)(xe * c - xo * s);
  Y[base + 1] = (_Float16)(xe * s + xo * c);
}

// V [S][KVD] fp32 -> Vt [KVD][S] f16 (so PV B-fragments load contiguously).
__global__ void __launch_bounds__(256) vtrans_f16(const float* __restrict__ V,
                                                  _Float16* __restrict__ Vt) {
  size_t gid = (size_t)blockIdx.x * 256 + threadIdx.x;  // over S*KVD
  int d = (int)(gid & (KVD_ - 1));
  int s = (int)(gid >> 10);
  Vt[(size_t)d * S_LEN + s] = (_Float16)V[gid];
}

// Flash-style causal attention. One wave per (head, 16-query block).
// Per 32-key step: 8 WMMA for Q*K^T (K-dim 128) + 8 WMMA for P*V.
__global__ void __launch_bounds__(256) attn_wmma(const _Float16* __restrict__ Qh,  // [S][DIM]
                                                 const _Float16* __restrict__ Kh,  // [S][KVD]
                                                 const _Float16* __restrict__ Vt,  // [KVD][S]
                                                 _Float16* __restrict__ Oh) {      // [S][DIM]
  __shared__ __align__(16) _Float16 smem[8][16][32];  // per-wave P staging (1 KB)
  const int lane = threadIdx.x & 31;
  const int wave = threadIdx.x >> 5;
  const int task = blockIdx.x * 8 + wave;   // NH_ * (S/16) tasks
  const int qb   = task & 127;              // waves in a block share head h
  const int h    = task >> 7;
  const int kv   = h >> 2;                  // GQA: 4 query heads per KV head
  const int q0   = qb * 16;
  const int nlo  = lane & 15;
  const int half = lane >> 4;

  v16h qf[4];
#pragma unroll
  for (int kk = 0; kk < 4; ++kk)
    qf[kk] = load_frag(Qh + h * HD_, q0, DIM_, kk * 32, lane);

  const v8f z = {0.f, 0.f, 0.f, 0.f, 0.f, 0.f, 0.f, 0.f};
  v8f o[8];
#pragma unroll
  for (int j = 0; j < 8; ++j) o[j] = z;
  float mrow[8], lrow[8];
#pragma unroll
  for (int r = 0; r < 8; ++r) { mrow[r] = -1e30f; lrow[r] = 0.f; }

  const float scale = 0.08838834764831845f;  // 1/sqrt(128)

  for (int kbase = 0; kbase <= q0 + 15; kbase += 32) {
    // ---- scores: S(16x32) = Q(16x128) x K_block^T ----
    v8f sc[2];
    sc[0] = z; sc[1] = z;
#pragma unroll
    for (int t = 0; t < 2; ++t)
#pragma unroll
      for (int kk = 0; kk < 4; ++kk) {
        v16h bf = load_frag(Kh + kv * HD_, kbase + t * 16, KVD_, kk * 32, lane);
        sc[t] = wmma16(qf[kk], bf, sc[t]);
      }
    // scale + causal mask (element coords from C-fragment layout)
#pragma unroll
    for (int t = 0; t < 2; ++t)
#pragma unroll
      for (int r = 0; r < 8; ++r) {
        float v  = sc[t][r] * scale;
        int   kg = kbase + t * 16 + nlo;
        int   qg = q0 + r + 8 * half;
        sc[t][r] = (kg > qg) ? -1e30f : v;
      }
    // row max: reduce across the 16 lanes of each half (xor 1..8 stays in-half)
    float tmax[8];
#pragma unroll
    for (int r = 0; r < 8; ++r) tmax[r] = fmaxf(sc[0][r], sc[1][r]);
#pragma unroll
    for (int off = 8; off >= 1; off >>= 1)
#pragma unroll
      for (int r = 0; r < 8; ++r)
        tmax[r] = fmaxf(tmax[r], __shfl_xor(tmax[r], off, 32));

    // online softmax: rescale running state, exponentiate, stage P in LDS
    float corr[8], psum[8];
#pragma unroll
    for (int r = 0; r < 8; ++r) {
      float mnew = fmaxf(mrow[r], tmax[r]);
      corr[r] = __expf(mrow[r] - mnew);
      mrow[r] = mnew;
      float e0 = __expf(sc[0][r] - mnew);
      float e1 = __expf(sc[1][r] - mnew);
      psum[r] = e0 + e1;
      int m = r + 8 * half;
      smem[wave][m][nlo]      = (_Float16)e0;
      smem[wave][m][16 + nlo] = (_Float16)e1;
    }
#pragma unroll
    for (int off = 8; off >= 1; off >>= 1)
#pragma unroll
      for (int r = 0; r < 8; ++r) psum[r] += __shfl_xor(psum[r], off, 32);
#pragma unroll
    for (int r = 0; r < 8; ++r) lrow[r] = lrow[r] * corr[r] + psum[r];
#pragma unroll
    for (int j = 0; j < 8; ++j)
#pragma unroll
      for (int r = 0; r < 8; ++r) o[j][r] *= corr[r];

    // LDS round-trip converts C-fragment layout -> A-fragment layout
    asm volatile("s_wait_dscnt 0" ::: "memory");
    Frag16 pf;
    pf.h[0] = *(const v8h*)(&smem[wave][nlo][half * 8]);
    pf.h[1] = *(const v8h*)(&smem[wave][nlo][16 + half * 8]);

    // ---- O(16x128) += P(16x32) x V_block(32x128), V transposed ----
#pragma unroll
    for (int j = 0; j < 8; ++j) {
      v16h bf = load_frag(Vt + (size_t)kv * HD_ * S_LEN, j * 16, S_LEN, kbase, lane);
      o[j] = wmma16(pf.v, bf, o[j]);
    }
  }

  // normalize (one reciprocal per row) and emit f16 attention output
#pragma unroll
  for (int r = 0; r < 8; ++r) {
    const float inv = 1.0f / lrow[r];
    const int m = q0 + r + 8 * half;
#pragma unroll
    for (int j = 0; j < 8; ++j) {
      int n = h * HD_ + j * 16 + nlo;
      Oh[(size_t)m * DIM_ + n] = (_Float16)(o[j][r] * inv);
    }
  }
}

extern "C" void kernel_launch(void* const* d_in, const int* in_sizes, int n_in,
                              void* d_out, int out_size, void* d_ws, size_t ws_size,
                              hipStream_t stream) {
  (void)in_sizes; (void)n_in; (void)out_size; (void)ws_size;
  const float* x  = (const float*)d_in[0];
  const float* wq = (const float*)d_in[1];
  const float* wk = (const float*)d_in[2];
  const float* wv = (const float*)d_in[3];
  const float* wo = (const float*)d_in[4];

  char* ws = (char*)d_ws;
  size_t off = 0;
  auto alloc = [&](size_t bytes) -> char* {
    char* p = ws + off;
    off += (bytes + 255) & ~(size_t)255;
    return p;
  };
  _Float16* xh  = (_Float16*)alloc((size_t)S_LEN * DIM_ * 2);
  _Float16* wqh = (_Float16*)alloc((size_t)DIM_ * DIM_ * 2);
  _Float16* wkh = (_Float16*)alloc((size_t)KVD_ * DIM_ * 2);
  _Float16* wvh = (_Float16*)alloc((size_t)KVD_ * DIM_ * 2);
  _Float16* woh = (_Float16*)alloc((size_t)DIM_ * DIM_ * 2);
  float*    Qf  = (float*)alloc((size_t)S_LEN * DIM_ * 4);
  float*    Kf  = (float*)alloc((size_t)S_LEN * KVD_ * 4);
  float*    Vf  = (float*)alloc((size_t)S_LEN * KVD_ * 4);
  _Float16* Qh  = (_Float16*)alloc((size_t)S_LEN * DIM_ * 2);
  _Float16* Khb = (_Float16*)alloc((size_t)S_LEN * KVD_ * 2);
  _Float16* Vt  = (_Float16*)alloc((size_t)KVD_ * S_LEN * 2);
  _Float16* Ah  = (_Float16*)alloc((size_t)S_LEN * DIM_ * 2);

  cvt_f32_f16<<<2048, 256, 0, stream>>>(x,  xh,  (size_t)S_LEN * DIM_);
  cvt_f32_f16<<<2048, 256, 0, stream>>>(wq, wqh, (size_t)DIM_ * DIM_);
  cvt_f32_f16<<<2048, 256, 0, stream>>>(wk, wkh, (size_t)KVD_ * DIM_);
  cvt_f32_f16<<<2048, 256, 0, stream>>>(wv, wvh, (size_t)KVD_ * DIM_);
  cvt_f32_f16<<<2048, 256, 0, stream>>>(wo, woh, (size_t)DIM_ * DIM_);

  // block tile 128(M) x 256(N)
  gemm_wmma_f16<DIM_, DIM_><<<dim3(DIM_ / 256, S_LEN / 128), 256, 0, stream>>>(xh, wqh, Qf);
  gemm_wmma_f16<KVD_, DIM_><<<dim3(KVD_ / 256, S_LEN / 128), 256, 0, stream>>>(xh, wkh, Kf);
  gemm_wmma_f16<KVD_, DIM_><<<dim3(KVD_ / 256, S_LEN / 128), 256, 0, stream>>>(xh, wvh, Vf);

  rope_f16<<<(S_LEN * NH_ * 64) / 256, 256, 0, stream>>>(Qf, Qh, NH_);
  rope_f16<<<(S_LEN * NKV_ * 64) / 256, 256, 0, stream>>>(Kf, Khb, NKV_);
  vtrans_f16<<<(S_LEN * KVD_) / 256, 256, 0, stream>>>(Vf, Vt);

  attn_wmma<<<(NH_ * (S_LEN / 16)) / 8, 256, 0, stream>>>(Qh, Khb, Vt, Ah);

  gemm_wmma_f16<DIM_, DIM_><<<dim3(DIM_ / 256, S_LEN / 128), 256, 0, stream>>>(Ah, woh, (float*)d_out);
}
